// S_SimDec_360777253446
// MI455X (gfx1250) — matrix operand order repaired and checked by reading, not compiled, not run on hardware
//
#include <hip/hip_runtime.h>

typedef __bf16 bf16_t;
typedef __attribute__((ext_vector_type(16))) __bf16 v16bf;
typedef __attribute__((ext_vector_type(8)))  __bf16 v8bf;
typedef __attribute__((ext_vector_type(8)))  float  v8f;

#define B_SZ 2048
#define H_SZ 256
#define G_SZ 5
#define T_SZ 64

// ---------------------------------------------------------------------------
// Generic bf16 WMMA GEMM:
//   OUT[M x N] (f32) = A0[M x K0] @ W0[N x K0]^T  (+ A1[M x K1] @ W1[N x K1]^T)
//                      + bias0[N] (+ bias1[N])
// A*, W* are bf16 row-major. M % 128 == 0, N % 64 == 0, K % 32 == 0.
// Block = 256 threads = 8 wave32 waves. Block tile = 128 (M) x 64 (N).
// Each wave computes a 16 x 64 strip with four v_wmma_f32_16x16x32_bf16 accs.
// W tile (64 cols x 32 K) is staged through LDS and shared by all 8 waves.
// ---------------------------------------------------------------------------
__global__ __launch_bounds__(256)
void gemm_bf16_wmma(const bf16_t* __restrict__ A0, const bf16_t* __restrict__ W0, int K0,
                    const bf16_t* __restrict__ A1, const bf16_t* __restrict__ W1, int K1,
                    const float* __restrict__ bias0, const float* __restrict__ bias1,
                    float* __restrict__ OUT, int M, int N)
{
    __shared__ bf16_t lB[64 * 32];  // 64 output-cols x 32 K, bf16 (4 KB)

    const int tid    = threadIdx.x;
    const int wave   = tid >> 5;
    const int lane   = tid & 31;
    const int laneLo = lane & 15;
    const int laneHi = lane >> 4;            // 0 or 1
    const int rowBase = blockIdx.x * 128 + wave * 16;
    const int colBase = blockIdx.y * 64;

    v8f acc[4];
#pragma unroll
    for (int j = 0; j < 4; ++j)
#pragma unroll
        for (int e = 0; e < 8; ++e) acc[j][e] = 0.0f;

#pragma unroll 1
    for (int pass = 0; pass < 2; ++pass) {
        const bf16_t* A = pass ? A1 : A0;
        const bf16_t* W = pass ? W1 : W0;
        const int     K = pass ? K1 : K0;
        if (A == nullptr || K == 0) continue;

        for (int k0 = 0; k0 < K; k0 += 32) {
            // --- cooperative stage of W tile into LDS ------------------------
            {
                const int n  = tid >> 2;         // 0..63 (output col within tile)
                const int ko = (tid & 3) * 8;    // 0,8,16,24
                const v8bf w8 = *(const v8bf*)(W + (size_t)(colBase + n) * K + k0 + ko);
                *(v8bf*)(&lB[n * 32 + ko]) = w8;
            }
            __syncthreads();

            // --- A fragment (16x32 bf16) per ISA lane layout -----------------
            const bf16_t* arow = A + (size_t)(rowBase + laneLo) * K + k0 + laneHi * 8;
            __builtin_prefetch(arow + 32, 0, 1);           // global_prefetch_b8
            const v8bf a0 = *(const v8bf*)(arow);          // K chunk  [0..8) (+8 hi-half)
            const v8bf a1 = *(const v8bf*)(arow + 16);     // K chunk [16..24) (+8 hi-half)
            const v16bf afrag = __builtin_shufflevector(
                a0, a1, 0, 1, 2, 3, 4, 5, 6, 7, 8, 9, 10, 11, 12, 13, 14, 15);

            // --- four 16x16 tiles along N ------------------------------------
#pragma unroll
            for (int j = 0; j < 4; ++j) {
                const bf16_t* bsrc = &lB[(j * 16 + laneLo) * 32 + laneHi * 8];
                const v8bf b0 = *(const v8bf*)(bsrc);
                const v8bf b1 = *(const v8bf*)(bsrc + 16);
                const v16bf bfrag = __builtin_shufflevector(
                    b0, b1, 0, 1, 2, 3, 4, 5, 6, 7, 8, 9, 10, 11, 12, 13, 14, 15);
                acc[j] = __builtin_amdgcn_wmma_f32_16x16x32_bf16(
                    false, afrag, false, bfrag, (short)0, acc[j], false, false);
            }
            __syncthreads();
        }
    }

    // --- epilogue: bias add + store (C layout: lane0-15 M=v, lane16-31 M=v+8) --
#pragma unroll
    for (int j = 0; j < 4; ++j) {
        const int col = colBase + j * 16 + laneLo;
        float badd = 0.0f;
        if (bias0) badd += bias0[col];
        if (bias1) badd += bias1[col];
#pragma unroll
        for (int v = 0; v < 8; ++v) {
            const int row = rowBase + v + laneHi * 8;
            if (row < M) OUT[(size_t)row * N + col] = acc[j][v] + badd;
        }
    }
}

// ---------------------------------------------------------------------------
// LSTM gate nonlinearity + state update. gates[B,1024] = (i,f,g,o) blocks.
// ---------------------------------------------------------------------------
__device__ __forceinline__ float sigf(float x) { return 1.0f / (1.0f + __expf(-x)); }

__global__ __launch_bounds__(256)
void lstm_pointwise(const float* __restrict__ gates, const float* __restrict__ c_prev,
                    float* __restrict__ c_out, float* __restrict__ h_f32,
                    bf16_t* __restrict__ h_bf, bf16_t* __restrict__ ycat, int ystride,
                    int rows)
{
    const int idx = blockIdx.x * blockDim.x + threadIdx.x;
    if (idx >= rows * H_SZ) return;
    const int b = idx >> 8, hh = idx & 255;
    const float* g = gates + (size_t)b * 1024;
    const float gi = sigf(g[hh]);
    const float gf = sigf(g[256 + hh]);
    const float gg = tanhf(g[512 + hh]);
    const float go = sigf(g[768 + hh]);
    const float c2 = gf * c_prev[idx] + gi * gg;
    const float h2 = go * tanhf(c2);
    c_out[idx] = c2;
    if (h_f32) h_f32[idx] = h2;
    h_bf[idx] = (bf16_t)h2;
    if (ycat) ycat[(size_t)b * ystride + hh] = (bf16_t)h2;
}

// ---------------------------------------------------------------------------
// Small utility kernels
// ---------------------------------------------------------------------------
__global__ void cvt_f32_bf16(const float* __restrict__ s, bf16_t* __restrict__ d, int n)
{
    const int i = blockIdx.x * blockDim.x + threadIdx.x;
    if (i < n) d[i] = (bf16_t)s[i];
}

__global__ void fill_f32(float* __restrict__ p, float v, int n)
{
    const int i = blockIdx.x * blockDim.x + threadIdx.x;
    if (i < n) p[i] = v;
}

__global__ void fill_bf16(bf16_t* __restrict__ p, int n)
{
    const int i = blockIdx.x * blockDim.x + threadIdx.x;
    if (i < n) p[i] = (bf16_t)0.0f;
}

__global__ void copy_or_add_f32(float* __restrict__ dst, const float* __restrict__ src,
                                int add, int n)
{
    const int i = blockIdx.x * blockDim.x + threadIdx.x;
    if (i < n) dst[i] = add ? (dst[i] + src[i]) : src[i];
}

__global__ void sos_fill(const float* __restrict__ emb, bf16_t* __restrict__ xt, int rows)
{
    const int idx = blockIdx.x * blockDim.x + threadIdx.x;
    if (idx < rows * H_SZ) xt[idx] = (bf16_t)emb[H_SZ + (idx & 255)];  // emb row 1 (SOS)
}

// ---------------------------------------------------------------------------
// Build combined [B,6,H] (bf16) from the tiny per-slot projections + embedding
// ---------------------------------------------------------------------------
__global__ __launch_bounds__(256)
void build_combined(const float* __restrict__ ci_all, const int* __restrict__ mode,
                    const float* c4pW, const float* c4pb, const float* c4oW, const float* c4ob,
                    const float* c4cW, const float* c4cb, const float* c4sW, const float* c4sb,
                    const float* poolW, const float* poolb, const float* emb,
                    bf16_t* __restrict__ comb)
{
    const int h  = threadIdx.x;
    const int bs = blockIdx.x;               // b*6 + slot
    const int b  = bs / 6, s = bs % 6;
    const float* ci = ci_all + (size_t)b * 28;
    float v;
    if (s == 0) { float a = c4pb[h]; for (int k = 0; k < 10; ++k) a += ci[k]      * c4pW[h * 10 + k]; v = a; }
    else if (s == 1) { float a = c4ob[h]; for (int k = 0; k < 8; ++k) a += ci[10 + k] * c4oW[h * 8 + k]; v = a; }
    else if (s == 2) { float a = c4cb[h]; for (int k = 0; k < 6; ++k) a += ci[18 + k] * c4cW[h * 6 + k]; v = a; }
    else if (s == 3) { float a = c4sb[h]; for (int k = 0; k < 4; ++k) a += ci[24 + k] * c4sW[h * 4 + k]; v = a; }
    else if (s == 4) { float m = 0.f; for (int k = 0; k < 28; ++k) m += ci[k]; m *= (1.0f / 28.0f);
                       v = m * poolW[h] + poolb[h]; }
    else             { v = emb[(size_t)mode[b] * H_SZ + h]; }
    comb[(size_t)bs * H_SZ + h] = (bf16_t)v;
}

// relu(fc_out [B*6,256]) -> time-major x [6,B,256] bf16
__global__ void relu_transpose(const float* __restrict__ fc, bf16_t* __restrict__ x, int B)
{
    const int idx = blockIdx.x * blockDim.x + threadIdx.x;
    const int r = idx >> 8, h = idx & 255;
    if (r >= B * 6) return;
    const int b = r / 6, s = r % 6;
    float v = fc[idx];
    v = v > 0.f ? v : 0.f;
    x[((size_t)s * B + b) * H_SZ + h] = (bf16_t)v;
}

// ---------------------------------------------------------------------------
// MDN head: logits/mu/sigma, log-softmax, softplus, hash-RNG sampling.
// Writes mus/sigmas/logpi into out at [3,T,B,G] and the sampled scalar.
// ---------------------------------------------------------------------------
__device__ __forceinline__ unsigned pcg_hash(unsigned s)
{
    s = s * 747796405u + 2891336453u;
    unsigned w = ((s >> ((s >> 28) + 4u)) ^ s) * 277803737u;
    return (w >> 22) ^ w;
}

__global__ __launch_bounds__(256)
void mdn_head(const float* __restrict__ dec_out,
              const float* piW, const float* pib,
              const float* muW, const float* mub,
              const float* sgW, const float* sgb,
              float* __restrict__ out, int t, float* __restrict__ sample, int B)
{
    const int b = blockIdx.x * blockDim.x + threadIdx.x;
    if (b >= B) return;
    const float* x = dec_out + (size_t)b * H_SZ;
    float lp[G_SZ], mu[G_SZ], sg[G_SZ];
#pragma unroll
    for (int g = 0; g < G_SZ; ++g) { lp[g] = pib[g]; mu[g] = mub[g]; sg[g] = sgb[g]; }
    for (int h = 0; h < H_SZ; ++h) {
        const float xv = x[h];
#pragma unroll
        for (int g = 0; g < G_SZ; ++g) {
            lp[g] += xv * piW[g * H_SZ + h];
            mu[g] += xv * muW[g * H_SZ + h];
            sg[g] += xv * sgW[g * H_SZ + h];
        }
    }
    // log-softmax (TEMP == 1)
    float m = lp[0];
#pragma unroll
    for (int g = 1; g < G_SZ; ++g) m = fmaxf(m, lp[g]);
    float s = 0.f;
#pragma unroll
    for (int g = 0; g < G_SZ; ++g) s += __expf(lp[g] - m);
    const float lse = m + __logf(s);
    float logpi[G_SZ], sig[G_SZ];
#pragma unroll
    for (int g = 0; g < G_SZ; ++g) {
        logpi[g] = lp[g] - lse;
        const float v = sg[g];
        sig[g] = (v > 20.f ? v : log1pf(__expf(v))) + 1e-3f;
    }
    // output layout [3, T, B, G]
    const size_t base = ((size_t)t * B + b) * G_SZ;
    const size_t blk  = (size_t)T_SZ * B * G_SZ;
#pragma unroll
    for (int g = 0; g < G_SZ; ++g) {
        out[base + g]           = mu[g];
        out[blk + base + g]     = sig[g];
        out[2 * blk + base + g] = logpi[g];
    }
    // categorical + Box-Muller sample (counter-based hash RNG; no state)
    const unsigned s0 = pcg_hash(((unsigned)t * 0x9E3779B9u) ^ (unsigned)b ^ 0xC0FFEE01u);
    const unsigned s1 = pcg_hash(s0 ^ 0xDEADBEEFu);
    const unsigned s2 = pcg_hash(s1 ^ 0x85EBCA6Bu);
    const float u1 = (s0 >> 8) * (1.0f / 16777216.0f);
    const float u2 = fmaxf((s1 >> 8) * (1.0f / 16777216.0f), 1e-7f);
    const float u3 = (s2 >> 8) * (1.0f / 16777216.0f);
    int comp = -1; float cdf = 0.f;
#pragma unroll
    for (int g = 0; g < G_SZ; ++g) {
        cdf += __expf(logpi[g]);
        if (comp < 0 && u1 <= cdf) comp = g;
    }
    if (comp < 0) comp = G_SZ - 1;
    const float nrm = sqrtf(-2.0f * __logf(u2)) * __cosf(6.2831853f * u3);
    sample[b] = mu[comp] + sig[comp] * nrm;
}

// nxt = ALPHA*(sample*sp_W + sp_b) + (1-ALPHA)*di_out -> bf16 feedback input
__global__ void next_input(const float* __restrict__ sample, const float* __restrict__ spW,
                           const float* __restrict__ spb, const float* __restrict__ diout,
                           bf16_t* __restrict__ xt, int B)
{
    const int idx = blockIdx.x * blockDim.x + threadIdx.x;
    if (idx >= B * H_SZ) return;
    const int b = idx >> 8, h = idx & 255;
    const float v = 0.8f * (sample[b] * spW[h] + spb[h]) + 0.2f * diout[idx];
    xt[idx] = (bf16_t)v;
}

// ---------------------------------------------------------------------------
// Host orchestration
// ---------------------------------------------------------------------------
extern "C" void kernel_launch(void* const* d_in, const int* in_sizes, int n_in,
                              void* d_out, int out_size, void* d_ws, size_t ws_size,
                              hipStream_t stream)
{
    (void)in_sizes; (void)n_in; (void)out_size; (void)ws_size;
    // input order: c_input, shipping_mode, tgt, then params in dict order
    const float* c_input = (const float*)d_in[0];
    const int*   mode    = (const int*)d_in[1];
    // d_in[2] = tgt (only its shape T is used by the reference)
    const float* P[50];
    for (int i = 3; i < 50; ++i) P[i] = (const float*)d_in[i];
    enum { I_C4PW = 3, I_C4PB, I_C4OW, I_C4OB, I_C4CW, I_C4CB, I_C4SW, I_C4SB,
           I_POOLW, I_POOLB, I_EMB, I_FCW, I_FCB, I_DIW, I_DIB, I_SPW, I_SPB,
           I_PIW, I_PIB, I_MUW, I_MUB, I_SGW, I_SGB,
           I_ENC0 = 26,   // enc{l}_{d}: Wih, Whh, bih, bhh  (l0f,l0b,l1f,l1b)
           I_DEC0 = 42 }; // dec{l}: Wih, Whh, bih, bhh

    // ----- workspace carve-out --------------------------------------------
    char* ws = (char*)d_ws;
    size_t off = 0;
    auto carve = [&](size_t bytes) -> void* {
        void* p = ws + off;
        off = (off + bytes + 255) & ~(size_t)255;
        return p;
    };
    bf16_t* fcWbf  = (bf16_t*)carve((size_t)H_SZ * H_SZ * 2);
    bf16_t* diWbf  = (bf16_t*)carve((size_t)H_SZ * H_SZ * 2);
    bf16_t* encWih[4]; bf16_t* encWhh[4];
    const int encK[4] = { 256, 256, 512, 512 };
    for (int i = 0; i < 4; ++i) encWih[i] = (bf16_t*)carve((size_t)1024 * encK[i] * 2);
    for (int i = 0; i < 4; ++i) encWhh[i] = (bf16_t*)carve((size_t)1024 * 256 * 2);
    bf16_t* decWih[2]; bf16_t* decWhh[2];
    for (int i = 0; i < 2; ++i) decWih[i] = (bf16_t*)carve((size_t)1024 * 256 * 2);
    for (int i = 0; i < 2; ++i) decWhh[i] = (bf16_t*)carve((size_t)1024 * 256 * 2);
    bf16_t* bufA   = (bf16_t*)carve((size_t)6 * B_SZ * 512 * 2);  // x / combined
    bf16_t* bufB   = (bf16_t*)carve((size_t)6 * B_SZ * 512 * 2);  // fc_out(f32) then y-concat
    float*  gates  = (float*)carve((size_t)B_SZ * 1024 * 4);
    float*  cbuf   = (float*)carve((size_t)B_SZ * H_SZ * 4);
    bf16_t* hbf    = (bf16_t*)carve((size_t)B_SZ * H_SZ * 2);
    float*  hfin   = (float*)carve((size_t)B_SZ * H_SZ * 4);
    float*  h0dec  = (float*)carve((size_t)2 * B_SZ * H_SZ * 4);
    float*  c0dec  = (float*)carve((size_t)2 * B_SZ * H_SZ * 4);  // becomes running dec c
    bf16_t* dhbf[2];
    dhbf[0] = (bf16_t*)carve((size_t)B_SZ * H_SZ * 2);
    dhbf[1] = (bf16_t*)carve((size_t)B_SZ * H_SZ * 2);
    float*  decout = (float*)carve((size_t)B_SZ * H_SZ * 4);
    bf16_t* xtbf   = (bf16_t*)carve((size_t)B_SZ * H_SZ * 2);
    float*  diout  = (float*)carve((size_t)B_SZ * H_SZ * 4);
    float*  sample = (float*)carve((size_t)B_SZ * 4);

    const int BH  = B_SZ * H_SZ;            // 524288
    const dim3 blk256(256);
    auto gcv = [](int n) { return dim3((n + 255) / 256); };

    // ----- one-time bf16 weight conversion --------------------------------
    cvt_f32_bf16<<<gcv(H_SZ * H_SZ), blk256, 0, stream>>>(P[I_FCW], fcWbf, H_SZ * H_SZ);
    cvt_f32_bf16<<<gcv(H_SZ * H_SZ), blk256, 0, stream>>>(P[I_DIW], diWbf, H_SZ * H_SZ);
    for (int i = 0; i < 4; ++i) {
        const int n = 1024 * encK[i];
        cvt_f32_bf16<<<gcv(n), blk256, 0, stream>>>(P[I_ENC0 + i * 4 + 0], encWih[i], n);
        cvt_f32_bf16<<<gcv(1024 * 256), blk256, 0, stream>>>(P[I_ENC0 + i * 4 + 1], encWhh[i],
                                                             1024 * 256);
    }
    for (int i = 0; i < 2; ++i) {
        cvt_f32_bf16<<<gcv(1024 * 256), blk256, 0, stream>>>(P[I_DEC0 + i * 4 + 0], decWih[i],
                                                             1024 * 256);
        cvt_f32_bf16<<<gcv(1024 * 256), blk256, 0, stream>>>(P[I_DEC0 + i * 4 + 1], decWhh[i],
                                                             1024 * 256);
    }

    // ----- build combined -> fc -> relu -> time-major x --------------------
    build_combined<<<dim3(B_SZ * 6), blk256, 0, stream>>>(
        c_input, mode, P[I_C4PW], P[I_C4PB], P[I_C4OW], P[I_C4OB], P[I_C4CW], P[I_C4CB],
        P[I_C4SW], P[I_C4SB], P[I_POOLW], P[I_POOLB], P[I_EMB], bufA);
    gemm_bf16_wmma<<<dim3((B_SZ * 6) / 128, H_SZ / 64), blk256, 0, stream>>>(
        bufA, fcWbf, 256, nullptr, nullptr, 0, P[I_FCB], nullptr,
        (float*)bufB, B_SZ * 6, H_SZ);
    relu_transpose<<<gcv(B_SZ * 6 * H_SZ), blk256, 0, stream>>>((float*)bufB, bufA, B_SZ);

    // ----- bidirectional 2-layer encoder over the 6 slots ------------------
    const bf16_t* inp = bufA;
    int inDim = 256;
    for (int l = 0; l < 2; ++l) {
        for (int d = 0; d < 2; ++d) {
            const int wi = l * 2 + d;
            const float* bih = P[I_ENC0 + wi * 4 + 2];
            const float* bhh = P[I_ENC0 + wi * 4 + 3];
            fill_f32<<<gcv(BH), blk256, 0, stream>>>(cbuf, 0.0f, BH);
            fill_bf16<<<gcv(BH), blk256, 0, stream>>>(hbf, BH);
            for (int step = 0; step < 6; ++step) {
                const int t = d ? (5 - step) : step;
                gemm_bf16_wmma<<<dim3(B_SZ / 128, 1024 / 64), blk256, 0, stream>>>(
                    inp + (size_t)t * B_SZ * inDim, encWih[wi], inDim,
                    hbf, encWhh[wi], 256, bih, bhh, gates, B_SZ, 1024);
                bf16_t* yc = (l == 0) ? (bufB + ((size_t)t * B_SZ) * 512 + d * 256) : nullptr;
                lstm_pointwise<<<gcv(BH), blk256, 0, stream>>>(
                    gates, cbuf, cbuf, (step == 5) ? hfin : nullptr, hbf, yc, 512, B_SZ);
            }
            copy_or_add_f32<<<gcv(BH), blk256, 0, stream>>>(h0dec + (size_t)l * BH, hfin, d, BH);
            copy_or_add_f32<<<gcv(BH), blk256, 0, stream>>>(c0dec + (size_t)l * BH, cbuf, d, BH);
        }
        inp = bufB;
        inDim = 512;
    }

    // ----- decoder init ----------------------------------------------------
    cvt_f32_bf16<<<gcv(BH), blk256, 0, stream>>>(h0dec, dhbf[0], BH);
    cvt_f32_bf16<<<gcv(BH), blk256, 0, stream>>>(h0dec + (size_t)BH, dhbf[1], BH);
    sos_fill<<<gcv(BH), blk256, 0, stream>>>(P[I_EMB], xtbf, B_SZ);

    // ----- 64 autoregressive MDN decoder steps -----------------------------
    for (int t = 0; t < T_SZ; ++t) {
        // layer 0: gates = xt@Wih0^T + h0@Whh0^T + b
        gemm_bf16_wmma<<<dim3(B_SZ / 128, 1024 / 64), blk256, 0, stream>>>(
            xtbf, decWih[0], 256, dhbf[0], decWhh[0], 256,
            P[I_DEC0 + 2], P[I_DEC0 + 3], gates, B_SZ, 1024);
        lstm_pointwise<<<gcv(BH), blk256, 0, stream>>>(
            gates, c0dec, c0dec, nullptr, dhbf[0], nullptr, 0, B_SZ);
        // layer 1
        gemm_bf16_wmma<<<dim3(B_SZ / 128, 1024 / 64), blk256, 0, stream>>>(
            dhbf[0], decWih[1], 256, dhbf[1], decWhh[1], 256,
            P[I_DEC0 + 6], P[I_DEC0 + 7], gates, B_SZ, 1024);
        lstm_pointwise<<<gcv(BH), blk256, 0, stream>>>(
            gates, c0dec + (size_t)BH, c0dec + (size_t)BH, decout, dhbf[1], nullptr, 0, B_SZ);
        // MDN head + sampling
        mdn_head<<<gcv(B_SZ), blk256, 0, stream>>>(
            decout, P[I_PIW], P[I_PIB], P[I_MUW], P[I_MUB], P[I_SGW], P[I_SGB],
            (float*)d_out, t, sample, B_SZ);
        // dec_in projection: di_out = dec_out @ di_W^T + di_b
        gemm_bf16_wmma<<<dim3(B_SZ / 128, H_SZ / 64), blk256, 0, stream>>>(
            dhbf[1], diWbf, 256, nullptr, nullptr, 0, P[I_DIB], nullptr,
            diout, B_SZ, H_SZ);
        // feedback mix -> next xt (bf16)
        next_input<<<gcv(BH), blk256, 0, stream>>>(sample, P[I_SPW], P[I_SPB], diout, xtbf, B_SZ);
    }
}
